// TriangleAttention_20186346291617
// MI455X (gfx1250) — matrix operand order repaired and checked by reading, not compiled, not run on hardware
//
#include <hip/hip_runtime.h>
#include <hip/hip_bf16.h>

typedef __attribute__((ext_vector_type(16))) _Float16 v16h;
typedef __attribute__((ext_vector_type(8)))  float    v8f;
typedef __attribute__((ext_vector_type(4)))  float    v4f;

#define NN   256
#define CZ   128
#define HNUM 4
#define DH   32
#define NPOS (NN * NN)   // 65536

// ---------- WMMA fragment loaders (contiguous-K layout: k = 16*(lane/16) + s for both A and B) ----------

// A-matrix 16x32 f16: lane L -> row L%16; 16 contiguous K [kb, kb+16), kb = 16*(L/16).
// rowp must point at (row-for-this-lane)*ld + col0 of a row-major f16 buffer.
__device__ __forceinline__ v16h load_frag_a(const _Float16* rowp) {
    int kb = ((threadIdx.x >> 4) & 1) * 16;
    union { int4 q[2]; v16h h; } u;
    u.q[0] = *(const int4*)(rowp + kb);
    u.q[1] = *(const int4*)(rowp + kb + 8);
    return u.h;
}

// B-matrix 32x16 f16: lane L -> column L%16; 16 contiguous K starting at 16*(L/16).
// p must already include the per-lane column-row base and the (lane>>4)*16 K offset.
__device__ __forceinline__ v16h load_frag_b(const _Float16* p) {
    union { int4 q[2]; v16h h; } u;
    u.q[0] = *(const int4*)(p);
    u.q[1] = *(const int4*)(p + 8);
    return u.h;
}

__device__ __forceinline__ v8f wmma_f16(v16h a, v16h b, v8f c) {
    return __builtin_amdgcn_wmma_f32_16x16x32_f16(false, a, false, b, (short)0, c, false, false);
}

// ---------- Kernel 0: weight f32 -> f16 (scale folded into wq) ----------
__global__ void convert_w_kernel(const float* wq, const float* wk, const float* wv,
                                 const float* wg, const float* wo, _Float16* Wh) {
    int id = blockIdx.x * blockDim.x + threadIdx.x;             // 0..16383
    Wh[id]             = (_Float16)(wq[id] * 0.17677669529663687f); // 1/sqrt(32)
    Wh[16384 + id]     = (_Float16)wk[id];
    Wh[2 * 16384 + id] = (_Float16)wv[id];
    Wh[3 * 16384 + id] = (_Float16)wg[id];
    Wh[4 * 16384 + id] = (_Float16)wo[id];
}

// ---------- Kernel 1: LayerNorm + triangle bias (1 wave per (q,k) position) ----------
__global__ void ln_tri_kernel(const float* __restrict__ x,
                              const float* __restrict__ ln_w, const float* __restrict__ ln_b,
                              const float* __restrict__ w_tri,
                              _Float16* __restrict__ xn, float* __restrict__ tri) {
    int tid  = blockIdx.x * blockDim.x + threadIdx.x;
    int pos  = tid >> 5;            // 0..65535
    int lane = tid & 31;
    const float4 a = ((const float4*)(x + (size_t)pos * CZ))[lane];   // channels lane*4..+3

    float s = a.x + a.y + a.z + a.w;
    #pragma unroll
    for (int off = 16; off; off >>= 1) s += __shfl_xor(s, off, 32);
    float mu = s * (1.0f / 128.0f);

    float d0 = a.x - mu, d1 = a.y - mu, d2 = a.z - mu, d3 = a.w - mu;
    float ss = d0 * d0 + d1 * d1 + d2 * d2 + d3 * d3;
    #pragma unroll
    for (int off = 16; off; off >>= 1) ss += __shfl_xor(ss, off, 32);
    float rstd = rsqrtf(ss * (1.0f / 128.0f) + 1e-5f);

    const float4 w4 = ((const float4*)ln_w)[lane];
    const float4 b4 = ((const float4*)ln_b)[lane];
    float y0 = d0 * rstd * w4.x + b4.x;
    float y1 = d1 * rstd * w4.y + b4.y;
    float y2 = d2 * rstd * w4.z + b4.z;
    float y3 = d3 * rstd * w4.w + b4.w;

    union { _Float16 e[4]; uint2 u; } o;
    o.e[0] = (_Float16)y0; o.e[1] = (_Float16)y1; o.e[2] = (_Float16)y2; o.e[3] = (_Float16)y3;
    *(uint2*)(xn + (size_t)pos * CZ + lane * 4) = o.u;

    // tri[h][q][k] = dot(xn[pos,:], w_tri[h,:])
    #pragma unroll
    for (int h = 0; h < HNUM; ++h) {
        const float4 wt = ((const float4*)(w_tri + h * CZ))[lane];
        float t = y0 * wt.x + y1 * wt.y + y2 * wt.z + y3 * wt.w;
        #pragma unroll
        for (int off = 16; off; off >>= 1) t += __shfl_xor(t, off, 32);
        if (lane == 0) tri[h * NPOS + pos] = t;
    }
}

// ---------- Kernel 2/5: rows x 128 GEMM vs 128x128 weight (B = W^T) ----------
// mode 0/1: plain f16 store (Q, K)   mode 2: transposed f16 store (Vt[i][e][k])
// mode 3: sigmoid(x + bias) f16 (gate)   mode 4: f32 store + bias, nontemporal (final out)
// Epilogue stages the block's 16x128 tile in LDS, then does coalesced 128-bit stores.
__global__ void gemm128_kernel(const _Float16* __restrict__ A, const _Float16* __restrict__ W,
                               _Float16* __restrict__ dst_h, float* __restrict__ dst_f,
                               const float* __restrict__ bias, int mode) {
    __shared__ __align__(16) char stbuf[16 * 128 * 4];            // 8KB staging
    _Float16 (*sh)[128] = (_Float16(*)[128])stbuf;                // [16 rows][128 e]
    float    (*sf)[128] = (float(*)[128])stbuf;                   // [16 rows][128 e]
    _Float16 (*sv)[16]  = (_Float16(*)[16])stbuf;                 // [128 e][16 k]

    int wv   = threadIdx.x >> 5;         // 0..7: e-tile
    int lane = threadIdx.x & 31;
    int rt   = blockIdx.x;               // row tile (16 rows)
    int row  = rt * 16 + (lane & 15);
    int co   = ((lane >> 4) & 1) * 16;   // B-fragment K offset within 32-chunk

    const _Float16* arow = A + (size_t)row * CZ;
    v8f acc = {};
    #pragma unroll
    for (int cc = 0; cc < 4; ++cc) {
        v16h af = load_frag_a(arow + cc * 32);
        const _Float16* bp = W + (size_t)(wv * 16 + (lane & 15)) * CZ + cc * 32 + co;
        v16h bf = load_frag_b(bp);
        acc = wmma_f16(af, bf, acc);
    }

    int n   = lane & 15;
    int mhi = (lane >> 4) * 8;
    #pragma unroll
    for (int v = 0; v < 8; ++v) {
        int m = v + mhi;
        int e = wv * 16 + n;
        float val = acc[v];
        if (mode == 4)      sf[m][e] = val + bias[e];
        else if (mode == 3) sh[m][e] = (_Float16)(1.0f / (1.0f + __expf(-(val + bias[e]))));
        else if (mode == 2) sv[e][m] = (_Float16)val;
        else                sh[m][e] = (_Float16)val;
    }
    __syncthreads();

    int t = threadIdx.x;
    if (mode == 4) {                       // 16x128 f32 tile -> two v4f NT stores/thread
        int r = t >> 4, c = (t & 15) * 8;
        float* dp = dst_f + (size_t)(rt * 16 + r) * CZ + c;
        __builtin_nontemporal_store(*(v4f*)&sf[r][c],     (v4f*)dp);
        __builtin_nontemporal_store(*(v4f*)&sf[r][c + 4], (v4f*)(dp + 4));
    } else if (mode == 2) {                // 128 e-rows x 16 k -> one int4/thread
        int e = t >> 1, part = t & 1;
        int i = rt >> 4;
        _Float16* dp = dst_h + (size_t)(i * CZ + e) * NN + (rt & 15) * 16 + part * 8;
        *(int4*)dp = *(int4*)&sv[e][part * 8];
    } else {                               // 16x128 f16 tile -> one int4/thread
        int r = t >> 4, c = (t & 15) * 8;
        *(int4*)(dst_h + (size_t)(rt * 16 + r) * CZ + c) = *(int4*)&sh[r][c];
    }
}

// ---------- Kernel 4: attention, one wave per (i, h, q-tile) ----------
#define AW 2   // waves per block (32KB LDS)
__global__ void attn_kernel(const _Float16* __restrict__ Qh, const _Float16* __restrict__ Kh,
                            const _Float16* __restrict__ Vt, const _Float16* __restrict__ Gh,
                            const float* __restrict__ tri, const float* __restrict__ mask,
                            _Float16* __restrict__ Og) {
    __shared__ float sc[AW][16][NN];     // per-wave 16x256 score/prob tile
    int wv   = threadIdx.x >> 5;
    int lane = threadIdx.x & 31;
    int task = blockIdx.x * AW + wv;     // 16384 tasks
    int qt = task & 15;
    int h  = (task >> 4) & 3;
    int i  = task >> 6;

    // Q A-fragment: rows qt*16..+16, d = 0..31 within head h
    int qrow = qt * 16 + (lane & 15);
    v16h qa = load_frag_a(Qh + ((size_t)(i * NN + qrow)) * CZ + h * DH);

    int n   = lane & 15;
    int mhi = (lane >> 4) * 8;

    // scores = Q K^T + mask_bias + tri
    #pragma unroll
    for (int kt = 0; kt < 16; ++kt) {
        int krow = kt * 16 + (lane & 15);
        v16h kb = load_frag_b(Kh + ((size_t)(i * NN + krow)) * CZ + h * DH + ((lane >> 4) & 1) * 16);
        v8f s = wmma_f16(qa, kb, (v8f){});
        int kg = kt * 16 + n;
        float mb = 1.0e9f * (mask[i * NN + kg] - 1.0f);
        const float* tp = tri + (size_t)h * NPOS + (size_t)(qt * 16 + mhi) * NN + kg;
        #pragma unroll
        for (int v = 0; v < 8; ++v)
            sc[wv][v + mhi][kg] = s[v] + mb + tp[(size_t)v * NN];
    }
    __syncthreads();

    // softmax over k (row-by-row, lane-parallel over columns)
    for (int r = 0; r < 16; ++r) {
        float vals[8];
        float mx = -1e30f;
        #pragma unroll
        for (int j = 0; j < 8; ++j) { vals[j] = sc[wv][r][lane + 32 * j]; mx = fmaxf(mx, vals[j]); }
        #pragma unroll
        for (int off = 16; off; off >>= 1) mx = fmaxf(mx, __shfl_xor(mx, off, 32));
        float sm = 0.0f;
        #pragma unroll
        for (int j = 0; j < 8; ++j) { vals[j] = __expf(vals[j] - mx); sm += vals[j]; }
        #pragma unroll
        for (int off = 16; off; off >>= 1) sm += __shfl_xor(sm, off, 32);
        float inv = 1.0f / sm;
        #pragma unroll
        for (int j = 0; j < 8; ++j) sc[wv][r][lane + 32 * j] = vals[j] * inv;
    }
    __syncthreads();

    // O = A @ V  (A rebuilt from LDS as f16 fragments; B from pre-transposed Vt)
    v8f acc0 = {}, acc1 = {};
    int m_a = lane & 15;
    int kb8 = (lane >> 4) * 16;
    #pragma unroll
    for (int kc = 0; kc < 8; ++kc) {
        union { _Float16 e[16]; v16h hh; } ua;
        #pragma unroll
        for (int t = 0; t < 8; ++t) ua.e[t]     = (_Float16)sc[wv][m_a][kc * 32 + kb8 + t];
        #pragma unroll
        for (int t = 0; t < 8; ++t) ua.e[8 + t] = (_Float16)sc[wv][m_a][kc * 32 + kb8 + 8 + t];
        const _Float16* vp0 = Vt + ((size_t)(i * CZ + h * DH + (lane & 15))) * NN
                                 + kc * 32 + ((lane >> 4) & 1) * 16;
        v16h b0 = load_frag_b(vp0);
        v16h b1 = load_frag_b(vp0 + (size_t)16 * NN);
        acc0 = wmma_f16(ua.hh, b0, acc0);
        acc1 = wmma_f16(ua.hh, b1, acc1);
    }

    // Stage the 16x32 output tile in this wave's own LDS region (same-wave DS ops are
    // in-order, no barrier needed), then gate with vector loads and do int4 stores.
    _Float16* ost = (_Float16*)&sc[wv][0][0];      // 512 halves = 1KB
    #pragma unroll
    for (int v = 0; v < 8; ++v) {
        int m = v + mhi;
        ost[m * 32 + n]      = (_Float16)acc0[v];
        ost[m * 32 + 16 + n] = (_Float16)acc1[v];
    }
    #pragma unroll
    for (int j = 0; j < 2; ++j) {
        int idx  = lane * 2 + j;                   // 0..63 -> (m, 8-half part)
        int m    = idx >> 2;
        int part = idx & 3;
        size_t go = ((size_t)(i * NN + qt * 16 + m)) * CZ + h * DH + part * 8;
        union { int4 q; _Float16 e[8]; } gv, ov;
        gv.q = *(const int4*)(Gh + go);
        ov.q = *(int4*)&ost[m * 32 + part * 8];
        #pragma unroll
        for (int t = 0; t < 8; ++t) ov.e[t] = (_Float16)((float)ov.e[t] * (float)gv.e[t]);
        *(int4*)(Og + go) = ov.q;
    }
}

extern "C" void kernel_launch(void* const* d_in, const int* in_sizes, int n_in,
                              void* d_out, int out_size, void* d_ws, size_t ws_size,
                              hipStream_t stream) {
    const float* x     = (const float*)d_in[0];
    const float* mask  = (const float*)d_in[1];
    const float* ln_w  = (const float*)d_in[2];
    const float* ln_b  = (const float*)d_in[3];
    const float* w_tri = (const float*)d_in[4];
    const float* wq    = (const float*)d_in[5];
    const float* wk    = (const float*)d_in[6];
    const float* wv    = (const float*)d_in[7];
    const float* wg    = (const float*)d_in[8];
    const float* bg    = (const float*)d_in[9];
    const float* wo    = (const float*)d_in[10];
    const float* bo    = (const float*)d_in[11];

    char* ws = (char*)d_ws;
    size_t off = 0;
    _Float16* Wh = (_Float16*)(ws + off); off += (size_t)5 * 16384 * 2; off = (off + 255) & ~(size_t)255;
    _Float16* xn = (_Float16*)(ws + off); off += (size_t)NPOS * CZ * 2;
    float*   tri = (float*)   (ws + off); off += (size_t)HNUM * NPOS * 4;
    _Float16* Qh = (_Float16*)(ws + off); off += (size_t)NPOS * CZ * 2;
    _Float16* Kh = (_Float16*)(ws + off); off += (size_t)NPOS * CZ * 2;
    _Float16* Vt = (_Float16*)(ws + off); off += (size_t)NPOS * CZ * 2;
    _Float16* Gh = (_Float16*)(ws + off); off += (size_t)NPOS * CZ * 2;
    _Float16* Og = (_Float16*)(ws + off); off += (size_t)NPOS * CZ * 2;

    convert_w_kernel<<<64, 256, 0, stream>>>(wq, wk, wv, wg, wo, Wh);
    ln_tri_kernel<<<NPOS / 8, 256, 0, stream>>>(x, ln_w, ln_b, w_tri, xn, tri);

    const int rowTiles = NPOS / 16;  // 4096
    gemm128_kernel<<<rowTiles, 256, 0, stream>>>(xn, Wh,              Qh, nullptr, bo, 0);
    gemm128_kernel<<<rowTiles, 256, 0, stream>>>(xn, Wh + 16384,      Kh, nullptr, bo, 1);
    gemm128_kernel<<<rowTiles, 256, 0, stream>>>(xn, Wh + 2 * 16384,  Vt, nullptr, bo, 2);
    gemm128_kernel<<<rowTiles, 256, 0, stream>>>(xn, Wh + 3 * 16384,  Gh, nullptr, bg, 3);

    attn_kernel<<<(NN * HNUM * 16) / AW, AW * 32, 0, stream>>>(Qh, Kh, Vt, Gh, tri, mask, Og);

    gemm128_kernel<<<rowTiles, 256, 0, stream>>>(Og, Wh + 4 * 16384, nullptr, (float*)d_out, bo, 4);
}
